// TGNN_22041772163566
// MI455X (gfx1250) — compile-verified, matrix operand order checked by reading, not run on hardware
//
#include <hip/hip_runtime.h>
#include <hip/hip_bf16.h>
#include <stdint.h>

// ---------------------------------------------------------------------------
// Problem dims (fixed by the reference)
// ---------------------------------------------------------------------------
#define N_NODES 20000
#define T_STEPS 8
#define C_DIM   256
#define H_DIM   512
#define E_EDGES 320000

typedef __attribute__((ext_vector_type(16))) __bf16   v16bf;
typedef __attribute__((ext_vector_type(8)))  float    v8f;
typedef __attribute__((ext_vector_type(4)))  uint32_t u32x4;
typedef __attribute__((ext_vector_type(8)))  uint32_t u32x8;

__device__ __forceinline__ float sigf(float x) { return 1.0f / (1.0f + __expf(-x)); }

// ---------------------------------------------------------------------------
// Weight packing kernels (fp32 -> bf16, GEMM-friendly layouts)
// ---------------------------------------------------------------------------
__global__ void pack_conv_w(const float* __restrict__ w, __bf16* __restrict__ out)
{
    // out[n][c] with c = tap*256 + in_ch ; w is (256,256,3) row-major
    int idx = blockIdx.x * blockDim.x + threadIdx.x;
    if (idx >= C_DIM * 3 * C_DIM) return;
    int n = idx / (3 * C_DIM);
    int c = idx - n * (3 * C_DIM);
    int s = c >> 8;          // tap
    int i = c & 255;         // input channel
    out[idx] = (__bf16)w[((size_t)n * C_DIM + i) * 3 + s];
}

__global__ void pack_concat2(const float* __restrict__ wl, const float* __restrict__ wr,
                             __bf16* __restrict__ out, int Kk, int total)
{
    int idx = blockIdx.x * blockDim.x + threadIdx.x;
    if (idx >= total) return;
    int n = idx / (2 * Kk);
    int c = idx - n * (2 * Kk);
    float v = (c < Kk) ? wl[(size_t)n * Kk + c] : wr[(size_t)n * Kk + (c - Kk)];
    out[idx] = (__bf16)v;
}

__global__ void vec_add(const float* __restrict__ a, const float* __restrict__ b,
                        float* __restrict__ o, int n)
{
    int i = blockIdx.x * blockDim.x + threadIdx.x;
    if (i < n) o[i] = a[i] + b[i];
}

// ---------------------------------------------------------------------------
// TDM: async-load one B panel (128 rows x 32 cols bf16, row stride Ktot elems)
// from global into LDS at lds_off, with TDM-applied padding so LDS rows land
// at 80-byte stride (== [row][40] bf16 layout). 2D tensor -> 2 SGPR groups.
// Issued per-wave (EXEC ignored); tracked on TENSORcnt.
// ---------------------------------------------------------------------------
__device__ __forceinline__ void tdm_load_b_panel(const __bf16* gptr, uint32_t lds_off,
                                                 int ktot)
{
    uint64_t ga = (uint64_t)(uintptr_t)gptr;
    u32x4 g0;
    g0[0] = 1u;                                   // count=1 (valid), user mode
    g0[1] = lds_off;                              // lds_addr (bytes)
    g0[2] = (uint32_t)ga;                         // global_addr[31:0]
    g0[3] = (uint32_t)((ga >> 32) & 0x01FFFFFFu)  // global_addr[56:32]
          | (2u << 30);                           // type = 2 ("image")
    u32x8 g1;
    g1[0] = (1u << 16)                            // data_size = 1 -> 2 bytes
          | (1u << 20)                            // pad_enable
          | (3u << 22)                            // pad_interval: 16 DWORDs (64B row)
          | (3u << 25);                           // pad_amount: 4 DWORDs (16B)
    g1[1] = (32u << 16);                          // tensor_dim0[15:0]  = 32
    g1[2] = (128u << 16);                         // dim0 hi=0 | tensor_dim1[15:0]=128
    g1[3] = (32u << 16);                          // dim1 hi=0 | tile_dim0 = 32
    g1[4] = 128u;                                 // tile_dim1 = 128, tile_dim2 = 0
    g1[5] = (uint32_t)ktot;                       // tensor_dim0_stride[31:0] (elems)
    g1[6] = 0u;                                   // stride0 hi | stride1 lo
    g1[7] = 0u;                                   // stride1 hi
    asm volatile("tensor_load_to_lds %0, %1" :: "s"(g0), "s"(g1) : "memory");
}

// ---------------------------------------------------------------------------
// Generic segmented bf16 WMMA GEMM (double-buffered LDS, TDM-staged B):
//   D(MxNn) = [A1 | A2 | A3](M x K1+K2+K3) @ B(Nn x Ktot).T + bias (+ReLU)
// Tile 64x128x32, 256 threads = 8 wave32 waves; each wave: 1 N-subtile,
// 4 M-subtiles -> 4x v_wmma_f32_16x16x32_bf16 per K chunk.
// ---------------------------------------------------------------------------
#define BM 64
#define BN 128
#define BK 32
#define LDSP 40   // padded LDS row stride (elements) == TDM pad layout

__global__ __launch_bounds__(256)
void gemm_seg_bf16(const float* __restrict__ A1, const float* __restrict__ A2,
                   const float* __restrict__ A3,
                   int K1, int K2, int K3,
                   const __bf16* __restrict__ B, const float* __restrict__ bias,
                   float* __restrict__ D, int M, int Nn, int act)
{
    __shared__ __bf16 sA[2][BM][LDSP];
    __shared__ __bf16 sB[2][BN][LDSP];

    const int tid  = threadIdx.x;
    const int lane = tid & 31;
    const int wv   = tid >> 5;        // 0..7 -> N subtile of this wave
    const int half = lane >> 4;       // 0/1 : lane group
    const int nl   = lane & 15;       // row (A) / col (B,C,D) within subtile

    const int m0 = blockIdx.x * BM;
    const int n0 = blockIdx.y * BN;
    const int Ktot = K1 + K2 + K3;
    const int nch  = Ktot / BK;

    // A staging coordinates: 8 contiguous fp32 per thread
    const int arow = tid >> 2;
    const int acb  = (tid & 3) * 8;
    const int gm   = m0 + arow;

    v8f acc[4] = {};

    // per-lane K offsets within one 32-wide chunk (16-bit A/B VGPR layout)
    int koff[8];
#pragma unroll
    for (int j = 0; j < 8; ++j)
        koff[j] = ((j >= 4) ? 16 : 0) + half * 8 + (j & 3) * 2;

    auto loadA = [&](int c, float (&v)[8]) {
        int k0 = c * BK;
        const float* As = A1; int kl = k0; int lda = K1;
        if (k0 >= K1)      { As = A2; kl = k0 - K1;      lda = K2; }
        if (k0 >= K1 + K2) { As = A3; kl = k0 - K1 - K2; lda = K3; }
        if (As != nullptr && gm < M) {
            const float* ap = As + (size_t)gm * lda + kl + acb;
#pragma unroll
            for (int j = 0; j < 8; ++j) v[j] = ap[j];
        } else {
#pragma unroll
            for (int j = 0; j < 8; ++j) v[j] = 0.0f;
        }
    };
    auto storeA = [&](int buf, const float (&v)[8]) {
#pragma unroll
        for (int j = 0; j < 8; ++j) sA[buf][arow][acb + j] = (__bf16)v[j];
    };

    // ---- prologue: stage chunk 0 into buffer 0 ----
    {
        float v[8];
        loadA(0, v);
        storeA(0, v);
        if (wv == 0) {
            tdm_load_b_panel(B + (size_t)n0 * Ktot,
                             (uint32_t)(uintptr_t)&sB[0][0][0], Ktot);
            __builtin_amdgcn_s_wait_tensorcnt(0);
        }
    }
    __syncthreads();

    int p = 0;
    for (int c = 0; c < nch; ++c) {
        float v[8];
        const bool more = (c + 1 < nch);
        if (more) {
            if (wv == 0)
                tdm_load_b_panel(B + (size_t)n0 * Ktot + (size_t)(c + 1) * BK,
                                 (uint32_t)(uintptr_t)&sB[p ^ 1][0][0], Ktot);
            loadA(c + 1, v);              // global loads overlap WMMA below
        }

        // ---- fragments from LDS buffer p, WMMA ----
        v16bf bfrag;
#pragma unroll
        for (int j = 0; j < 8; ++j) {
            bfrag[2 * j]     = sB[p][wv * 16 + nl][koff[j]];
            bfrag[2 * j + 1] = sB[p][wv * 16 + nl][koff[j] + 1];
        }
#pragma unroll
        for (int ms = 0; ms < 4; ++ms) {
            v16bf afrag;
#pragma unroll
            for (int j = 0; j < 8; ++j) {
                afrag[2 * j]     = sA[p][ms * 16 + nl][koff[j]];
                afrag[2 * j + 1] = sA[p][ms * 16 + nl][koff[j] + 1];
            }
            acc[ms] = __builtin_amdgcn_wmma_f32_16x16x32_bf16(
                false, afrag, false, bfrag, (short)0, acc[ms], false, false);
        }

        if (more) {
            storeA(p ^ 1, v);                         // commit next A panel
            if (wv == 0) __builtin_amdgcn_s_wait_tensorcnt(0);  // next B panel landed
        }
        __syncthreads();
        p ^= 1;
    }

    // ---- epilogue: C/D layout -> global, + bias, optional ReLU ----
    const int n  = n0 + wv * 16 + nl;
    const float bv = bias[n];
#pragma unroll
    for (int ms = 0; ms < 4; ++ms) {
#pragma unroll
        for (int r = 0; r < 8; ++r) {
            int m = m0 + ms * 16 + half * 8 + r;
            if (m < M) {
                float vv = acc[ms][r] + bv;
                if (act == 1) vv = fmaxf(vv, 0.0f);
                D[(size_t)m * Nn + n] = vv;
            }
        }
    }
}

// ---------------------------------------------------------------------------
// Graph aggregation: scatter-add of x[src] rows into acc[dst] (fp32 atomics),
// degree count, then in-place divide by max(count,1).
// ---------------------------------------------------------------------------
__global__ void scatter_add4(const float* __restrict__ x, const int* __restrict__ src,
                             const int* __restrict__ dst, float* __restrict__ acc, int C)
{
    int idx = blockIdx.x * blockDim.x + threadIdx.x;
    int per = C >> 2;                 // float4 slots per edge
    int e = idx / per;
    if (e >= E_EDGES) return;
    int c = (idx - e * per) << 2;
    int s = src[e], d = dst[e];
    const float4 v = *(const float4*)(x + (size_t)s * C + c);
    float* a = acc + (size_t)d * C + c;
    atomicAdd(a + 0, v.x);
    atomicAdd(a + 1, v.y);
    atomicAdd(a + 2, v.z);
    atomicAdd(a + 3, v.w);
}

__global__ void count_edges(const int* __restrict__ dst, float* __restrict__ cnt)
{
    int e = blockIdx.x * blockDim.x + threadIdx.x;
    if (e < E_EDGES) atomicAdd(&cnt[dst[e]], 1.0f);
}

__global__ void mean_div(float* __restrict__ acc, const float* __restrict__ cnt,
                         int C, int total)
{
    int idx = blockIdx.x * blockDim.x + threadIdx.x;
    if (idx < total) acc[idx] /= fmaxf(cnt[idx / C], 1.0f);
}

// ---------------------------------------------------------------------------
// LSTM pointwise gates: g (N,4H) -> update c, write h
// ---------------------------------------------------------------------------
__global__ void lstm_point(const float* __restrict__ g, float* __restrict__ c,
                           float* __restrict__ hout)
{
    int idx = blockIdx.x * blockDim.x + threadIdx.x;
    if (idx >= N_NODES * H_DIM) return;
    int n = idx >> 9;                 // H_DIM == 512
    int j = idx & (H_DIM - 1);
    const float* gr = g + (size_t)n * 4 * H_DIM;
    float iv = sigf(gr[j]);
    float fv = sigf(gr[H_DIM + j]);
    float gv = tanhf(gr[2 * H_DIM + j]);
    float ov = sigf(gr[3 * H_DIM + j]);
    float cn = fv * c[idx] + iv * gv;
    c[idx] = cn;
    hout[idx] = ov * tanhf(cn);
}

// ---------------------------------------------------------------------------
// Host orchestration
// ---------------------------------------------------------------------------
extern "C" void kernel_launch(void* const* d_in, const int* in_sizes, int n_in,
                              void* d_out, int out_size, void* d_ws, size_t ws_size,
                              hipStream_t stream)
{
    (void)in_sizes; (void)n_in; (void)out_size; (void)ws_size;

    const float* x      = (const float*)d_in[0];
    const int*   edges  = (const int*)d_in[1];
    const float* conv_w = (const float*)d_in[2];
    const float* conv_b = (const float*)d_in[3];
    const float* w1l    = (const float*)d_in[4];
    const float* b1l    = (const float*)d_in[5];
    const float* w1r    = (const float*)d_in[6];
    const float* w2l    = (const float*)d_in[7];
    const float* b2l    = (const float*)d_in[8];
    const float* w2r    = (const float*)d_in[9];
    const float* w_ih   = (const float*)d_in[10];
    const float* w_hh   = (const float*)d_in[11];
    const float* b_ih   = (const float*)d_in[12];
    const float* b_hh   = (const float*)d_in[13];

    char* ws = (char*)d_ws;
    size_t off = 0;
    auto take = [&](size_t bytes) -> char* {
        char* p = ws + off;
        off = (off + bytes + 255) & ~(size_t)255;
        return p;
    };

    __bf16* wBconv = (__bf16*)take((size_t)C_DIM * 3 * C_DIM * sizeof(__bf16));
    __bf16* wB1    = (__bf16*)take((size_t)H_DIM * 2 * C_DIM * sizeof(__bf16));
    __bf16* wB2    = (__bf16*)take((size_t)H_DIM * 2 * H_DIM * sizeof(__bf16));
    __bf16* wBL    = (__bf16*)take((size_t)4 * H_DIM * 2 * H_DIM * sizeof(__bf16));
    float*  bL     = (float*)take((size_t)4 * H_DIM * sizeof(float));
    float*  enh    = (float*)take((size_t)T_STEPS * N_NODES * C_DIM * sizeof(float));
    float*  aggr   = (float*)take((size_t)N_NODES * H_DIM * sizeof(float));
    float*  cnt    = (float*)take((size_t)N_NODES * sizeof(float));
    float*  h1     = (float*)take((size_t)N_NODES * H_DIM * sizeof(float));
    float*  nf     = (float*)take((size_t)T_STEPS * N_NODES * H_DIM * sizeof(float));
    float*  gbuf   = (float*)take((size_t)N_NODES * 4 * H_DIM * sizeof(float));
    float*  hbuf   = (float*)take((size_t)N_NODES * H_DIM * sizeof(float));
    float*  cbuf   = (float*)take((size_t)N_NODES * H_DIM * sizeof(float));

    const dim3 blk(256);
    const int MB = (N_NODES + BM - 1) / BM;    // 313 M-tiles

    // ---- pack weights to bf16 ----
    pack_conv_w<<<(C_DIM * 3 * C_DIM + 255) / 256, blk, 0, stream>>>(conv_w, wBconv);
    pack_concat2<<<(H_DIM * 2 * C_DIM + 255) / 256, blk, 0, stream>>>(
        w1l, w1r, wB1, C_DIM, H_DIM * 2 * C_DIM);
    pack_concat2<<<(H_DIM * 2 * H_DIM + 255) / 256, blk, 0, stream>>>(
        w2l, w2r, wB2, H_DIM, H_DIM * 2 * H_DIM);
    pack_concat2<<<(4 * H_DIM * 2 * H_DIM + 255) / 256, blk, 0, stream>>>(
        w_ih, w_hh, wBL, H_DIM, 4 * H_DIM * 2 * H_DIM);
    vec_add<<<(4 * H_DIM + 255) / 256, blk, 0, stream>>>(b_ih, b_hh, bL, 4 * H_DIM);

    // ---- phase 1: causal temporal conv as 3-segment GEMM ----
    for (int t = 0; t < T_STEPS; ++t) {
        const float* a1 = (t - 2 >= 0) ? x + (size_t)(t - 2) * N_NODES * C_DIM : nullptr;
        const float* a2 = (t - 1 >= 0) ? x + (size_t)(t - 1) * N_NODES * C_DIM : nullptr;
        const float* a3 = x + (size_t)t * N_NODES * C_DIM;
        gemm_seg_bf16<<<dim3(MB, C_DIM / BN), blk, 0, stream>>>(
            a1, a2, a3, C_DIM, C_DIM, C_DIM, wBconv, conv_b,
            enh + (size_t)t * N_NODES * C_DIM, N_NODES, C_DIM, 0);
    }

    // ---- phase 2: 2-layer SAGE per timestep ----
    for (int t = 0; t < T_STEPS; ++t) {
        const int* srcp = edges + (size_t)t * 2 * E_EDGES;
        const int* dstp = srcp + E_EDGES;
        const float* xt = enh + (size_t)t * N_NODES * C_DIM;

        // layer 1 aggregate (C=256)
        hipMemsetAsync(aggr, 0, (size_t)N_NODES * C_DIM * sizeof(float), stream);
        hipMemsetAsync(cnt, 0, (size_t)N_NODES * sizeof(float), stream);
        scatter_add4<<<((size_t)E_EDGES * (C_DIM / 4) + 255) / 256, blk, 0, stream>>>(
            xt, srcp, dstp, aggr, C_DIM);
        count_edges<<<(E_EDGES + 255) / 256, blk, 0, stream>>>(dstp, cnt);
        mean_div<<<((size_t)N_NODES * C_DIM + 255) / 256, blk, 0, stream>>>(
            aggr, cnt, C_DIM, N_NODES * C_DIM);
        // h1 = relu(mean @ w1l.T + xt @ w1r.T + b1l)
        gemm_seg_bf16<<<dim3(MB, H_DIM / BN), blk, 0, stream>>>(
            aggr, xt, nullptr, C_DIM, C_DIM, 0, wB1, b1l, h1, N_NODES, H_DIM, 1);

        // layer 2 aggregate (C=512)
        hipMemsetAsync(aggr, 0, (size_t)N_NODES * H_DIM * sizeof(float), stream);
        hipMemsetAsync(cnt, 0, (size_t)N_NODES * sizeof(float), stream);
        scatter_add4<<<((size_t)E_EDGES * (H_DIM / 4) + 255) / 256, blk, 0, stream>>>(
            h1, srcp, dstp, aggr, H_DIM);
        count_edges<<<(E_EDGES + 255) / 256, blk, 0, stream>>>(dstp, cnt);
        mean_div<<<((size_t)N_NODES * H_DIM + 255) / 256, blk, 0, stream>>>(
            aggr, cnt, H_DIM, N_NODES * H_DIM);
        // nf[t] = mean @ w2l.T + h1 @ w2r.T + b2l
        gemm_seg_bf16<<<dim3(MB, H_DIM / BN), blk, 0, stream>>>(
            aggr, h1, nullptr, H_DIM, H_DIM, 0, wB2, b2l,
            nf + (size_t)t * N_NODES * H_DIM, N_NODES, H_DIM, 0);
    }

    // ---- phase 3: LSTM scan ----
    hipMemsetAsync(hbuf, 0, (size_t)N_NODES * H_DIM * sizeof(float), stream);
    hipMemsetAsync(cbuf, 0, (size_t)N_NODES * H_DIM * sizeof(float), stream);
    for (int t = 0; t < T_STEPS; ++t) {
        // g = nf[t] @ w_ih.T + h @ w_hh.T + (b_ih + b_hh)
        gemm_seg_bf16<<<dim3(MB, (4 * H_DIM) / BN), blk, 0, stream>>>(
            nf + (size_t)t * N_NODES * H_DIM, hbuf, nullptr, H_DIM, H_DIM, 0,
            wBL, bL, gbuf, N_NODES, 4 * H_DIM, 0);
        float* hout = (t == T_STEPS - 1) ? (float*)d_out : hbuf;
        lstm_point<<<((size_t)N_NODES * H_DIM + 255) / 256, blk, 0, stream>>>(
            gbuf, cbuf, hout);
    }
}
